// PriorLayer_47081431499231
// MI455X (gfx1250) — compile-verified, matrix operand order checked by reading, not run on hardware
//
#include <hip/hip_runtime.h>

typedef __attribute__((ext_vector_type(16))) _Float16 v16h;
typedef __attribute__((ext_vector_type(8)))  float    v8f;

#define DIM 256
#define EPS 1e-10f

// Dynamic LDS layout (bytes):
//   Afrag : 131072  (256x256 f16 transition matrix in WMMA A-fragment order)
//   sfrag :    512  (256 f16 state vector in WMMA B-fragment order)
//   part  :    512  (32 x 4 f32 partial sums)
//   scal  :     16  (broadcast 1/S)
#define SMEM_BYTES (131072 + 512 + 512 + 16)

__global__ __launch_bounds__(512, 1)
void prior_layer_kernel(const float* __restrict__ ps,
                        const float* __restrict__ trans,
                        const float* __restrict__ sprior,
                        const float* __restrict__ bins,
                        float* __restrict__ out, int T)
{
    extern __shared__ __align__(64) unsigned char smem[];
    _Float16* Afrag = (_Float16*)smem;                       // 65536 halves
    _Float16* sfrag = (_Float16*)(smem + 131072);            // 256 halves
    float*    part  = (float*)   (smem + 131072 + 512);      // 32*4 floats
    float*    scal  = (float*)   (smem + 131072 + 1024);     // invS

    const int  tid    = threadIdx.x;
    const int  w      = tid >> 5;        // wave id: 0..15, owns row block w
    const int  lane   = tid & 31;
    const int  hi     = lane >> 4;       // lane group within wave
    const bool writer = (lane & 15) == 0; // lanes 0 and 16 own 8 outputs each

    // ---- one-time: swizzle f32 transition matrix -> f16 A-fragment layout ----
    // A tile (i,j) is 16x32; lane g*16+m holds 16 halves:
    //   g=0: K = {0..7,16..23}, g=1: K = {8..15,24..31}   (ISA 16-bit A layout)
    for (int e = tid; e < DIM * DIM; e += 512) {
        int r = e >> 8, k = e & 255;
        int i = r >> 4, mrow = r & 15;
        int j = k >> 5, kk = k & 31;
        int g  = (kk >> 3) & 1;
        int ln = g * 16 + mrow;
        int q  = (kk & 7) | ((kk & 16) >> 1);
        Afrag[(((i * 8 + j) * 32 + ln) << 4) + q] = (_Float16)trans[e];
    }
    // ---- one-time: state prior -> B-fragment order ----
    // per K-block j the 32 values are stored [0..7,16..23,8..15,24..31] so each
    // lane group reads one contiguous 32B v16h (state replicated across columns)
    if (tid < DIM) {
        int m = tid, j = m >> 5, kk = m & 31;
        int grp = kk >> 3;
        int pg  = ((grp & 1) << 1) | (grp >> 1);   // swap middle groups
        sfrag[j * 32 + pg * 8 + (kk & 7)] = (_Float16)sprior[m];
    }
    __syncthreads();

    // ---- per-writer-lane constants / observation double buffer ----
    const int mb = w * 16 + hi * 8;      // base output index of this writer lane
    float breg[8];
    float4 pc0 = {}, pc1 = {};
    if (writer) {
        const float4* bq = (const float4*)(bins + mb);
        float4 b0 = bq[0], b1 = bq[1];
        breg[0]=b0.x; breg[1]=b0.y; breg[2]=b0.z; breg[3]=b0.w;
        breg[4]=b1.x; breg[5]=b1.y; breg[6]=b1.z; breg[7]=b1.w;
        const float4* pq = (const float4*)(ps + mb);   // t = 0
        pc0 = pq[0]; pc1 = pq[1];
    }
    // destination slot in sfrag for this writer lane's normalized chunk
    const int jd  = mb >> 5, kb = mb & 31;
    const int gpd = kb >> 3;
    _Float16* sdst = sfrag + jd * 32 + (((gpd & 1) << 1) | (gpd >> 1)) * 8;

    const v16h* Ap = ((const v16h*)Afrag) + (w * 8 * 32 + lane);
    const v16h* Bp = ((const v16h*)sfrag) + hi;

    for (int t = 0; t < T; ++t) {
        // consume current observation, immediately prefetch the next one so the
        // global load latency hides under the WMMA chain
        float pcur[8];
        if (writer) {
            pcur[0]=pc0.x; pcur[1]=pc0.y; pcur[2]=pc0.z; pcur[3]=pc0.w;
            pcur[4]=pc1.x; pcur[5]=pc1.y; pcur[6]=pc1.z; pcur[7]=pc1.w;
            int tn = (t + 1 < T) ? (t + 1) : t;
            const float4* pq = (const float4*)(ps + (size_t)tn * DIM + mb);
            pc0 = pq[0]; pc1 = pq[1];
            if (t + 8 < T)
                __builtin_prefetch(ps + (size_t)(t + 8) * DIM + mb, 0, 0);
        }

        // ---- y = Trans @ state : 8 chained v_wmma_f32_16x16x32_f16 ----
        // state replicated across B columns -> every D column holds y block
        v8f acc = {};
#pragma unroll
        for (int j = 0; j < 8; ++j) {
            v16h a = Ap[j * 32];   // 2x ds_load_b128, conflict-free fragment
            v16h b = Bp[j * 2];    // broadcast within lane group
            acc = __builtin_amdgcn_wmma_f32_16x16x32_f16(
                false, a, false, b, (short)0, acc, false, false);
        }

        // ---- elementwise + local partial reductions (writer lanes) ----
        float pn[8];
        if (writer) {
            float sS = 0.f, sB = 0.f, sL = 0.f;
#pragma unroll
            for (int q = 0; q < 8; ++q) {
                float y = acc[q];                       // lane0: M=q, lane16: M=8+q
                float v = __builtin_fmaf(y, pcur[q], EPS);
                pn[q] = v;
                sS += v;
                sB = __builtin_fmaf(v, breg[q], sB);
                sL = __builtin_fmaf(v, __logf(v), sL);
            }
            int slot = (w * 2 + hi) * 4;
            part[slot + 0] = sS;
            part[slot + 1] = sB;
            part[slot + 2] = sL;
        }
        __syncthreads();

        // ---- cross-wave reduction in wave 0 (5-step shfl_xor tree) ----
        if (w == 0) {
            int slot = lane * 4;
            float rS = part[slot + 0];
            float rB = part[slot + 1];
            float rL = part[slot + 2];
#pragma unroll
            for (int off = 16; off >= 1; off >>= 1) {
                rS += __shfl_xor(rS, off, 32);
                rB += __shfl_xor(rB, off, 32);
                rL += __shfl_xor(rL, off, 32);
            }
            if (lane == 0) {
                float invS = 1.0f / rS;
                out[t]     = rB * invS;                    // E_x[t]
                out[T + t] = __logf(rS) - rL * invS;       // entropy[t]
                scal[0]    = invS;
            }
        }
        __syncthreads();

        // ---- normalize and write state back in B-fragment order ----
        if (writer) {
            float invS = scal[0];
#pragma unroll
            for (int q = 0; q < 8; ++q)
                sdst[q] = (_Float16)(pn[q] * invS);
        }
        __syncthreads();
    }
}

extern "C" void kernel_launch(void* const* d_in, const int* in_sizes, int n_in,
                              void* d_out, int out_size, void* d_ws, size_t ws_size,
                              hipStream_t stream) {
    const float* ps     = (const float*)d_in[0];  // (T, 256)
    const float* trans  = (const float*)d_in[1];  // (256, 256)
    const float* sprior = (const float*)d_in[2];  // (256,)
    const float* bins   = (const float*)d_in[3];  // (256,)
    float* out = (float*)d_out;                   // E_x (T) then uncert (T)
    int T = in_sizes[0] / DIM;

    (void)d_ws; (void)ws_size; (void)n_in; (void)out_size;

    // 132 KB dynamic LDS (WGP has 320 KB) — opt in above the default cap
    hipFuncSetAttribute((const void*)prior_layer_kernel,
                        hipFuncAttributeMaxDynamicSharedMemorySize, SMEM_BYTES);

    prior_layer_kernel<<<dim3(1), dim3(512), SMEM_BYTES, stream>>>(
        ps, trans, sprior, bins, out, T);
}
